// MLPMoE_29171417875051
// MI455X (gfx1250) — compile-verified, matrix operand order checked by reading, not run on hardware
//
#include <hip/hip_runtime.h>
#include <hip/hip_bf16.h>

// Problem constants (from reference): B=2, S=1024 -> T=2048 tokens
#define T_TOK 2048
#define DDIM  1024
#define IDIM  1024
#define NEXP  8
#define BM    32          // assignments (rows) per expert block
#define GLIMIT 7.0f
#define GALPHA 1.702f

typedef __attribute__((ext_vector_type(16))) __bf16 v16bf;
typedef __attribute__((ext_vector_type(8)))  float  v8f;

__device__ __forceinline__ __bf16 us2bf(unsigned short u) {
    __bf16 r; __builtin_memcpy(&r, &u, 2); return r;
}
// float -> bf16 bits, round-to-nearest-even
__device__ __forceinline__ unsigned short f2bf_us(float f) {
    unsigned u = __builtin_bit_cast(unsigned, f);
    u += 0x7fffu + ((u >> 16) & 1u);
    return (unsigned short)(u >> 16);
}

// A fragment (16x32 bf16), row-major bf16 source: lane L -> row M=L%16,
// K pattern k = 32*kt + 2*(v&3) + 8*(L>>4) + 16*(v>>2)  (ISA 16-bit A layout)
__device__ __forceinline__ v16bf load_a_bf16(const unsigned short* row, int kt, int hf) {
    v16bf a;
#pragma unroll
    for (int v = 0; v < 8; ++v) {
        const int koff = kt * 32 + ((v & 3) << 1) + (hf << 3) + ((v >> 2) << 4);
        const unsigned ua = *(const unsigned*)(row + koff);
        a[2 * v]     = us2bf((unsigned short)(ua & 0xffffu));
        a[2 * v + 1] = us2bf((unsigned short)(ua >> 16));
    }
    return a;
}
// B fragment (32x16 bf16) from row-major f32 [N][K] weights, converted in-flight
__device__ __forceinline__ v16bf load_b_f32(const float* row, int kt, int hf) {
    v16bf b;
#pragma unroll
    for (int v = 0; v < 8; ++v) {
        const int koff = kt * 32 + ((v & 3) << 1) + (hf << 3) + ((v >> 2) << 4);
        const float2 wp = *(const float2*)(row + koff);
        b[2 * v]     = (__bf16)wp.x;
        b[2 * v + 1] = (__bf16)wp.y;
    }
    return b;
}
#define WMMA_BF16(A, B, C) __builtin_amdgcn_wmma_f32_16x16x32_bf16( \
    false, (A), false, (B), (short)0, (C), false, false)

// ---------------------------------------------------------------------------
// Kernel 0: zero expert counters
// ---------------------------------------------------------------------------
__global__ void zero_cnt_kernel(int* __restrict__ cnt) {
    if (threadIdx.x < NEXP) cnt[threadIdx.x] = 0;
}

// ---------------------------------------------------------------------------
// Kernel 1: RMSNorm + gating + top-2 routing.
// ---------------------------------------------------------------------------
__global__ void __launch_bounds__(256)
norm_gate_kernel(const float* __restrict__ x, const float* __restrict__ norm_w,
                 const float* __restrict__ gate_w, const float* __restrict__ gate_b,
                 unsigned short* __restrict__ hb, int* __restrict__ cnt,
                 int* __restrict__ a_tok, float* __restrict__ a_wt) {
    __shared__ float red[256];
    __shared__ float lg[NEXP];
    const int t = blockIdx.x;
    const int tid = threadIdx.x;
    const float* xr = x + (size_t)t * DDIM;

    float xs[4];
    float ss = 0.f;
#pragma unroll
    for (int i = 0; i < 4; ++i) { xs[i] = xr[tid + 256 * i]; ss += xs[i] * xs[i]; }
    red[tid] = ss;
    __syncthreads();
    for (int s = 128; s > 0; s >>= 1) {
        if (tid < s) red[tid] += red[tid + s];
        __syncthreads();
    }
    const float rms = rsqrtf(red[0] * (1.0f / (float)DDIM) + 1.1920929e-7f);
    if (tid < NEXP) lg[tid] = 0.f;
    __syncthreads();

    float part[NEXP];
#pragma unroll
    for (int e = 0; e < NEXP; ++e) part[e] = 0.f;
#pragma unroll
    for (int i = 0; i < 4; ++i) {
        const int d = tid + 256 * i;
        const float h = xs[i] * rms * norm_w[d];
        hb[(size_t)t * DDIM + d] = f2bf_us(h);
#pragma unroll
        for (int e = 0; e < NEXP; ++e) part[e] += h * gate_w[e * DDIM + d];
    }
#pragma unroll
    for (int e = 0; e < NEXP; ++e) atomicAdd(&lg[e], part[e]);  // ds_add_f32
    __syncthreads();

    if (tid == 0) {
        float v0 = -1e30f, v1 = -1e30f;
        int e0 = 0, e1 = 0;
#pragma unroll
        for (int e = 0; e < NEXP; ++e) {
            const float v = lg[e] + gate_b[e];
            if (v > v0) { v1 = v0; e1 = e0; v0 = v; e0 = e; }
            else if (v > v1) { v1 = v; e1 = e; }
        }
        const float w0 = 1.f / (1.f + __expf(v1 - v0));  // softmax, v0 >= v1
        const float w1 = 1.f - w0;
        int p0 = atomicAdd(&cnt[e0], 1);
        a_tok[e0 * T_TOK + p0] = t * 2 + 0;
        a_wt [e0 * T_TOK + p0] = w0;
        int p1 = atomicAdd(&cnt[e1], 1);
        a_tok[e1 * T_TOK + p1] = t * 2 + 1;
        a_wt [e1 * T_TOK + p1] = w1;
    }
}

// ---------------------------------------------------------------------------
// Kernel 2: per-expert fused MLP1 + SwiGLU + MLP2, bf16 WMMA, f32 accumulate.
// grid = (T_TOK/BM tiles, NEXP); block = 256 threads = 8 wave32.
// BM=32: two 16-row tiles share every B fragment. MLP1 computes separate
// glu (even cols) / lin (odd cols) tiles so SwiGLU is pure per-lane VALU.
// Dynamic LDS 128KB (320KB/WGP -> 2 blocks/WGP).
// ---------------------------------------------------------------------------
__global__ void __launch_bounds__(256)
moe_expert_kernel(const unsigned short* __restrict__ hb,
                  const int* __restrict__ cnt,
                  const int* __restrict__ a_tok, const float* __restrict__ a_wt,
                  const float* __restrict__ w1, const float* __restrict__ b1,
                  const float* __restrict__ w2, const float* __restrict__ b2,
                  float* __restrict__ contrib) {
    extern __shared__ __align__(16) unsigned short smem[];
    unsigned short* hS   = smem;               // BM x DDIM bf16 = 64 KB
    unsigned short* actS = smem + BM * DDIM;   // BM x IDIM bf16 = 64 KB
    __shared__ int   tokS[BM];
    __shared__ float wtS[BM];
    __shared__ int   neS;

    const int e    = blockIdx.y;
    const int base = blockIdx.x * BM;
    const int tid  = threadIdx.x;
    const int n_e  = cnt[e];
    if (base >= n_e) return;                   // uniform per block

    if (tid < BM) {
        const int idx = base + tid;
        const bool valid = idx < n_e;
        tokS[tid] = valid ? a_tok[e * T_TOK + idx] : 0;
        wtS[tid]  = valid ? a_wt [e * T_TOK + idx] : 0.f;
    }
    if (tid == 0) neS = n_e;
    __syncthreads();

    // Stage BM gathered bf16 activation rows into LDS (read from L2 once).
    {
        const int r  = tid >> 3;               // 32 rows, 8 threads per row
        const int c8 = tid & 7;
        const uint4* src = (const uint4*)(hb + (size_t)(tokS[r] >> 1) * DDIM);
        uint4* dst = (uint4*)(hS + r * DDIM);
#pragma unroll
        for (int j = 0; j < 16; ++j)           // 128 uint4 per row / 8 threads
            dst[c8 + j * 8] = src[c8 + j * 8];
    }
    __syncthreads();

    const int wv   = tid >> 5;                 // wave id 0..7
    const int lane = tid & 31;
    const int nloc = lane & 15;
    const int hf   = lane >> 4;

    // ---------------- MLP1 (D -> 2I) + clamped SwiGLU ----------------
    const float* w1e = w1 + (size_t)e * (2 * IDIM) * DDIM;
    const unsigned short* h0 = hS + nloc * DDIM;          // A rows, tile mt=0
    const unsigned short* h1 = hS + (16 + nloc) * DDIM;   // tile mt=1
    for (int nt = 0; nt < 8; ++nt) {
        const int c0 = wv * 256 + nt * 32;     // 32 interleaved t1 columns
        const int cg = c0 + 2 * nloc;          // this lane's glu column (even)
        const float biasG = b1[(size_t)e * 2 * IDIM + cg];
        const float biasL = b1[(size_t)e * 2 * IDIM + cg + 1];
        v8f accG[2], accL[2];
#pragma unroll
        for (int v = 0; v < 8; ++v) {
            accG[0][v] = biasG; accG[1][v] = biasG;
            accL[0][v] = biasL; accL[1][v] = biasL;
        }
        const float* wrowG = w1e + (size_t)cg * DDIM;     // row-major [2I][D]
        const float* wrowL = wrowG + DDIM;
#pragma unroll 2
        for (int kt = 0; kt < DDIM / 32; ++kt) {
            const v16bf bg = load_b_f32(wrowG, kt, hf);
            const v16bf bl = load_b_f32(wrowL, kt, hf);
            const v16bf a0 = load_a_bf16(h0, kt, hf);
            const v16bf a1 = load_a_bf16(h1, kt, hf);
            accG[0] = WMMA_BF16(a0, bg, accG[0]);
            accG[1] = WMMA_BF16(a1, bg, accG[1]);
            accL[0] = WMMA_BF16(a0, bl, accL[0]);
            accL[1] = WMMA_BF16(a1, bl, accL[1]);
        }
        // SwiGLU: glu & lin now in the same lane/VGPR slot — pure VALU
        const int icol = (c0 >> 1) + nloc;
#pragma unroll
        for (int mt = 0; mt < 2; ++mt) {
#pragma unroll
            for (int v = 0; v < 8; ++v) {
                const float g = fminf(accG[mt][v], GLIMIT);
                const float l = fminf(fmaxf(accL[mt][v], -GLIMIT), GLIMIT);
                const float sg = __builtin_amdgcn_rcpf(1.f + __expf(-GALPHA * g));
                const float r = g * sg * (l + 1.f);
                const int M = mt * 16 + v + (hf << 3);
                actS[M * IDIM + icol] = f2bf_us(r);
            }
        }
    }
    __syncthreads();

    // ---------------- MLP2 (I -> D) + weighted scatter ----------------
    const float* w2e = w2 + (size_t)e * DDIM * IDIM;
    const unsigned short* ar0 = actS + nloc * IDIM;
    const unsigned short* ar1 = actS + (16 + nloc) * IDIM;
    for (int nt = 0; nt < 8; ++nt) {
        const int d0 = wv * 128 + nt * 16;
        const int d  = d0 + nloc;
        const float bias = b2[(size_t)e * DDIM + d];
        v8f acc[2];
#pragma unroll
        for (int v = 0; v < 8; ++v) { acc[0][v] = bias; acc[1][v] = bias; }
        const float* wrow = w2e + (size_t)d * IDIM;       // row-major [D][I]
#pragma unroll 2
        for (int kt = 0; kt < IDIM / 32; ++kt) {
            const v16bf b  = load_b_f32(wrow, kt, hf);
            const v16bf a0 = load_a_bf16(ar0, kt, hf);
            const v16bf a1 = load_a_bf16(ar1, kt, hf);
            acc[0] = WMMA_BF16(a0, b, acc[0]);
            acc[1] = WMMA_BF16(a1, b, acc[1]);
        }
#pragma unroll
        for (int mt = 0; mt < 2; ++mt) {
#pragma unroll
            for (int v = 0; v < 8; ++v) {
                const int M = mt * 16 + v + (hf << 3);
                if (base + M < neS) {
                    const int pv = tokS[M];               // token*2 + slot
                    contrib[(size_t)pv * DDIM + d] = acc[mt][v] * wtS[M];
                }
            }
        }
    }
}

// ---------------------------------------------------------------------------
// Kernel 3: out = x + contrib[t, slot0] + contrib[t, slot1]   (float4)
// ---------------------------------------------------------------------------
__global__ void combine_kernel(const float4* __restrict__ x,
                               const float4* __restrict__ contrib,
                               float4* __restrict__ out) {
    const size_t i = (size_t)blockIdx.x * 256 + threadIdx.x;   // over T*D/4
    const size_t dq = DDIM / 4;
    const size_t t = i / dq;
    const size_t d = i - t * dq;
    const float4 a  = x[i];
    const float4 c0 = contrib[(t * 2) * dq + d];
    const float4 c1 = contrib[(t * 2 + 1) * dq + d];
    out[i] = make_float4(a.x + c0.x + c1.x, a.y + c0.y + c1.y,
                         a.z + c0.z + c1.z, a.w + c0.w + c1.w);
}

// ---------------------------------------------------------------------------
extern "C" void kernel_launch(void* const* d_in, const int* in_sizes, int n_in,
                              void* d_out, int out_size, void* d_ws, size_t ws_size,
                              hipStream_t stream) {
    const float* x      = (const float*)d_in[0];
    const float* norm_w = (const float*)d_in[1];
    const float* gate_w = (const float*)d_in[2];
    const float* gate_b = (const float*)d_in[3];
    const float* mlp1_w = (const float*)d_in[4];
    const float* mlp1_b = (const float*)d_in[5];
    const float* mlp2_w = (const float*)d_in[6];
    const float* mlp2_b = (const float*)d_in[7];
    float* out = (float*)d_out;

    // Workspace layout (~20.3 MB total)
    char* ws = (char*)d_ws;
    size_t off = 0;
    unsigned short* hb = (unsigned short*)(ws + off); off += (size_t)T_TOK * DDIM * 2;
    int*   cnt     = (int*)(ws + off);   off += 256;
    int*   a_tok   = (int*)(ws + off);   off += (size_t)NEXP * T_TOK * 4;
    float* a_wt    = (float*)(ws + off); off += (size_t)NEXP * T_TOK * 4;
    float* contrib = (float*)(ws + off);

    const int dynlds = 2 * BM * DDIM * 2;  // 128 KB (hS + actS)
    (void)hipFuncSetAttribute((const void*)moe_expert_kernel,
                              hipFuncAttributeMaxDynamicSharedMemorySize, dynlds);

    zero_cnt_kernel<<<1, 32, 0, stream>>>(cnt);
    norm_gate_kernel<<<T_TOK, 256, 0, stream>>>(x, norm_w, gate_w, gate_b,
                                                hb, cnt, a_tok, a_wt);
    dim3 g2(T_TOK / BM, NEXP);
    moe_expert_kernel<<<g2, 256, dynlds, stream>>>(hb, cnt, a_tok, a_wt,
                                                   mlp1_w, mlp1_b, mlp2_w, mlp2_b,
                                                   contrib);
    combine_kernel<<<(T_TOK * DDIM / 4) / 256, 256, 0, stream>>>(
        (const float4*)x, (const float4*)contrib, (float4*)out);
}
